// Retriever_88545045774999
// MI455X (gfx1250) — compile-verified, matrix operand order checked by reading, not run on hardware
//
#include <hip/hip_runtime.h>
#include <hip/hip_bf16.h>
#include <stdint.h>

#define DEV __device__ __forceinline__

typedef __bf16 bf16_t;
typedef bf16_t v16bf __attribute__((ext_vector_type(16)));
typedef float  v8f   __attribute__((ext_vector_type(8)));

struct Blk32 { int4 lo, hi; };

// ---- bf16 helpers (round-to-nearest-even) ------------------------------
DEV unsigned rne_bf16_bits(float f) {
    unsigned u = __float_as_uint(f);
    return (u + 0x7FFFu + ((u >> 16) & 1u)) >> 16;
}
DEV unsigned pk2(float a, float b) {
    unsigned ua = __float_as_uint(a); ua = ua + 0x7FFFu + ((ua >> 16) & 1u);
    unsigned ub = __float_as_uint(b); ub = ub + 0x7FFFu + ((ub >> 16) & 1u);
    return (ua >> 16) | (ub & 0xFFFF0000u);
}
DEV v8f vzero() { v8f z = {0.f,0.f,0.f,0.f,0.f,0.f,0.f,0.f}; return z; }

// A fragment (16x32 bf16, M x K) from an LDS strip stored row-major.
// lane L: row m = L&15; half = L>>4 selects K sub-block per ISA layout:
//   VGPR0-3: K = half*8 + [0..7]; VGPR4-7: K = 16 + half*8 + [0..7]
DEV v16bf fragA_lds(const uint16_t* strip, int strideE, int m, int hf, int kc) {
    const char* p = (const char*)strip + (size_t)(m * strideE) * 2 + kc * 64 + hf * 16;
    Blk32 t; t.lo = *(const int4*)p; t.hi = *(const int4*)(p + 32);
    return __builtin_bit_cast(v16bf, t);
}
// B fragment (32x16 bf16, K x N) from prepacked global buffer:
// layout [frag][lane][8 dwords] -> one 32B coalesced load per lane.
DEV v16bf fragB_g(const unsigned* buf, int frag, int lane) {
    const int4* q = (const int4*)(buf + ((size_t)frag * 32 + lane) * 8);
    Blk32 t; t.lo = q[0]; t.hi = q[1];
    return __builtin_bit_cast(v16bf, t);
}
DEV v8f wmma_bf16(v16bf a, v16bf b, v8f c) {
    return __builtin_amdgcn_wmma_f32_16x16x32_bf16(false, a, false, b, (short)0, c, false, false);
}

// ======================= helper kernels =================================
__global__ void k_zero(float* p, long n) {
    long i = (long)blockIdx.x * blockDim.x + threadIdx.x;
    if (i < n) p[i] = 0.f;
}
// f32 -> bf16 table conversion (entity/relation/non-text embeddings)
__global__ void k_cvt(const float* __restrict__ s, uint16_t* d, long n) {
    long i = (long)blockIdx.x * blockDim.x + threadIdx.x;
    if (i < n) d[i] = (uint16_t)rne_bf16_bits(s[i]);
}
__global__ void k_count(int E, const int* __restrict__ hid, const int* __restrict__ tid_,
                        float* cin, float* cou) {
    int i = blockIdx.x * blockDim.x + threadIdx.x;
    if (i < E) { atomicAdd(&cin[tid_[i]], 1.f); atomicAdd(&cou[hid[i]], 1.f); }
}
__global__ void k_initpos(int N, const float* __restrict__ topic, float* pos, float* cur) {
    int i = blockIdx.x * blockDim.x + threadIdx.x;
    if (i < N) {
        float a = topic[2 * i], b = topic[2 * i + 1];
        pos[(size_t)i * 10 + 0] = a; pos[(size_t)i * 10 + 1] = b;
        cur[2 * i] = a; cur[2 * i + 1] = b;
    }
}
__global__ void k_copy2(int N, const float* __restrict__ topic, float* cur) {
    int i = blockIdx.x * blockDim.x + threadIdx.x;
    if (i < 2 * N) cur[i] = topic[i];
}
__global__ void k_scatter(int E, const float* __restrict__ cur,
                          const int* __restrict__ src, const int* __restrict__ dst, float* acc) {
    int i = blockIdx.x * blockDim.x + threadIdx.x;
    if (i < E) {
        int s = src[i], d = dst[i];
        atomicAdd(&acc[2 * d + 0], cur[2 * s + 0]);
        atomicAdd(&acc[2 * d + 1], cur[2 * s + 1]);
    }
}
__global__ void k_norm(int N, const float* __restrict__ acc, const float* __restrict__ cnt,
                       float* nxt, float* pos, int col) {
    int i = blockIdx.x * blockDim.x + threadIdx.x;
    if (i < N) {
        float c = fmaxf(cnt[i], 1.f);
        float v0 = acc[2 * i] / c, v1 = acc[2 * i + 1] / c;
        nxt[2 * i] = v0; nxt[2 * i + 1] = v1;
        pos[(size_t)i * 10 + col] = v0; pos[(size_t)i * 10 + col + 1] = v1;
    }
}
// motif[e][d] = sum_l (ids!=0) wts[e,l]*motif_emb[ids[e,l]][d], stored bf16
__global__ void k_motif(long total, const int* __restrict__ ids, const float* __restrict__ wts,
                        const float* __restrict__ me, uint16_t* out) {
    long idx = (long)blockIdx.x * blockDim.x + threadIdx.x;
    if (idx >= total) return;
    long e = idx >> 6; int d = (int)(idx & 63);
    float a = 0.f;
#pragma unroll
    for (int l = 0; l < 8; ++l) {
        int id = ids[e * 8 + l];
        if (id != 0) a += wts[e * 8 + l] * me[(size_t)id * 64 + d];
    }
    out[idx] = (uint16_t)rne_bf16_bits(a);
}
// Prepack W [K x 256] (row-major f32) into per-lane bf16 B fragments.
__global__ void k_prepack(const float* __restrict__ W, int Kvalid, int kcCount, unsigned* dst) {
    int idx = blockIdx.x * blockDim.x + threadIdx.x;
    if (idx >= kcCount * 512) return;
    int lane = idx & 31, nt = (idx >> 5) & 15, kc = idx >> 9;
    int n = nt * 16 + (lane & 15);
    int kb = kc * 32 + (lane >> 4) * 8;
    unsigned* o = dst + (size_t)idx * 8;
#pragma unroll
    for (int j = 0; j < 8; ++j) {
        int k = kb + ((j < 4) ? 2 * j : 16 + 2 * (j - 4));
        float a = (k < Kvalid) ? W[(size_t)k * 256 + n] : 0.f;
        float b = (k + 1 < Kvalid) ? W[(size_t)(k + 1) * 256 + n] : 0.f;
        o[j] = pk2(a, b);
    }
}
// gate = softmax(q@W_gate + b_gate); b1c = b1 + q @ W1[:256,:]   (hoisted)
__global__ void k_gate_c1(const float* __restrict__ q, const float* __restrict__ W1,
                          const float* __restrict__ b1, const float* __restrict__ Wg,
                          const float* __restrict__ bg, float* b1c, float* gate) {
    __shared__ float lg[3];
    int t = threadIdx.x;
    if (t < 3) {
        float s = bg[t];
        for (int k = 0; k < 256; ++k) s += q[k] * Wg[k * 3 + t];
        lg[t] = s;
    }
    float c = 0.f;
    for (int k = 0; k < 256; ++k) c += q[k] * W1[(size_t)k * 256 + t];
    b1c[t] = b1[t] + c;
    __syncthreads();
    if (t == 0) {
        float mx = fmaxf(lg[0], fmaxf(lg[1], lg[2]));
        float e0 = expf(lg[0] - mx), e1 = expf(lg[1] - mx), e2 = expf(lg[2] - mx);
        float s = e0 + e1 + e2;
        gate[0] = e0 / s; gate[1] = e1 / s; gate[2] = e2 / s;
    }
}

// ======================= main fused edge kernel =========================
// 2 waves / block, 4 M-tiles (64 edges) / block.
//  - Each B fragment is loaded once per wave and feeds 4 WMMAs (M-reuse).
//  - Waves split the 16 N-tiles (wave0: nt 0-7, wave1: nt 8-15), so each
//    B fragment is loaded by exactly one wave (N-reuse across waves).
//  - Activation tables pre-converted to bf16: staging is pure b128 copies.
// Dynamic LDS (146,944 B):
//   sA1: 4 x 16 x 520  [h_h | h_t]      K=512 (+pad)
//   sA2: 4 x 16 x 328  [h_r | motif]    K=320 (+pad)
//   sAp: 4 x 16 x 32   [pos_h|pos_t|0]  K=32
//   sA3: 4 x 16 x 264  h_fused          K=256 (+pad)
//   red: 2 x 4 x 16 f32 cross-wave reduction buffer
#define ST1 520
#define ST2 328
#define STP 32
#define ST3 264
#define OFF_A2  (4 * 16 * ST1)
#define OFF_AP  (OFF_A2 + 4 * 16 * ST2)
#define OFF_A3  (OFF_AP + 4 * 16 * STP)
#define OFF_RED (OFF_A3 + 4 * 16 * ST3)
#define SMEM_ELTS (OFF_RED + 256)   // 128 floats = 256 u16 elts

__global__ __launch_bounds__(64)
void k_edges(int E, int n_text,
             const int* __restrict__ hid, const int* __restrict__ tid_, const int* __restrict__ rid,
             const uint16_t* __restrict__ entb, const uint16_t* __restrict__ ntxb,
             const uint16_t* __restrict__ relb,
             const uint16_t* __restrict__ motif, const float* __restrict__ pos,
             const float* __restrict__ gate,
             const float* __restrict__ b_nb, const float* __restrict__ b_pos,
             const float* __restrict__ b_str, const float* __restrict__ b1c,
             const float* __restrict__ W2, const float* __restrict__ b2,
             const unsigned* __restrict__ pWnb, const unsigned* __restrict__ pWstr,
             const unsigned* __restrict__ pWpos, const unsigned* __restrict__ pW1b,
             float* __restrict__ out) {
    extern __shared__ __align__(16) uint16_t smem[];
    uint16_t* sA1 = smem;
    uint16_t* sA2 = smem + OFF_A2;
    uint16_t* sAp = smem + OFF_AP;
    uint16_t* sA3 = smem + OFF_A3;
    float*    red = (float*)(smem + OFF_RED);

    const int tidx = threadIdx.x;
    const int lane = tidx & 31, w = tidx >> 5;
    const int m = lane & 15, hf = lane >> 4;
    const int base = blockIdx.x * 64;

    // this wave stages M-tiles 2w and 2w+1
    const int t0 = 2 * w, t1 = 2 * w + 1;
    int ea = base + t0 * 16 + m;  if (ea >= E) ea = E - 1;
    int eb = base + t1 * 16 + m;  if (eb >= E) eb = E - 1;
    const int ha = hid[ea], ta = tid_[ea];
    const int hb = hid[eb], tb = tid_[eb];
    const uint16_t* pha = (ha < n_text) ? entb + (size_t)ha * 256 : ntxb;
    const uint16_t* pta = (ta < n_text) ? entb + (size_t)ta * 256 : ntxb;
    const uint16_t* phb = (hb < n_text) ? entb + (size_t)hb * 256 : ntxb;
    const uint16_t* ptb = (tb < n_text) ? entb + (size_t)tb * 256 : ntxb;
    __builtin_prefetch(pha, 0, 3);                 // global_prefetch_b8
    __builtin_prefetch(pta, 0, 3);
    __builtin_prefetch(phb, 0, 3);
    __builtin_prefetch(ptb, 0, 3);

    // ---- stage A1 for this wave's two tiles (bf16 row copies) ----------
#pragma unroll
    for (int tt = 0; tt < 2; ++tt) {
        const uint16_t* s = tt ? (hf ? ptb : phb) : (hf ? pta : pha);
        const int ti = tt ? t1 : t0;
        char* dst = (char*)sA1 + (size_t)(ti * 16 * ST1 + m * ST1 + hf * 256) * 2;
#pragma unroll 8
        for (int c = 0; c < 256; c += 8) {
            int4 v = *(const int4*)((const char*)s + (size_t)c * 2);
            *(int4*)(dst + (size_t)c * 2) = v;
        }
    }
    // ---- stage A2: h_r (bf16 copy) + motif (bf16 copy) -----------------
#pragma unroll
    for (int tt = 0; tt < 2; ++tt) {
        const int et = tt ? eb : ea;
        const int ti = tt ? t1 : t0;
        if (hf == 0) {
            const uint16_t* s = relb + (size_t)rid[et] * 256;
            char* dst = (char*)sA2 + (size_t)(ti * 16 * ST2 + m * ST2) * 2;
#pragma unroll 8
            for (int c = 0; c < 256; c += 8) {
                int4 v = *(const int4*)((const char*)s + (size_t)c * 2);
                *(int4*)(dst + (size_t)c * 2) = v;
            }
        } else {
            const int4* s = (const int4*)(motif + (size_t)et * 64);
            int4* dst = (int4*)((char*)sA2 + (size_t)(ti * 16 * ST2 + m * ST2 + 256) * 2);
#pragma unroll
            for (int j = 0; j < 8; ++j) dst[j] = s[j];
        }
    }
    // ---- stage Apos: half 0 -> tile t0, half 1 -> tile t1 --------------
    {
        const int hh = hf ? hb : ha, th = hf ? tb : ta;
        const int ti = hf ? t1 : t0;
        const float* a = pos + (size_t)hh * 10;
        const float* b = pos + (size_t)th * 10;
        unsigned d[16];
        d[0] = pk2(a[0], a[1]); d[1] = pk2(a[2], a[3]); d[2] = pk2(a[4], a[5]);
        d[3] = pk2(a[6], a[7]); d[4] = pk2(a[8], a[9]);
        d[5] = pk2(b[0], b[1]); d[6] = pk2(b[2], b[3]); d[7] = pk2(b[4], b[5]);
        d[8] = pk2(b[6], b[7]); d[9] = pk2(b[8], b[9]);
#pragma unroll
        for (int j = 10; j < 16; ++j) d[j] = 0u;
        int4* dst = (int4*)((char*)sAp + ((size_t)ti * 16 * STP + (size_t)m * STP) * 2);
#pragma unroll
        for (int j = 0; j < 4; ++j) {
            int4 o; o.x = (int)d[4 * j]; o.y = (int)d[4 * j + 1];
            o.z = (int)d[4 * j + 2]; o.w = (int)d[4 * j + 3];
            dst[j] = o;
        }
    }
    __syncthreads();

    const float g0 = gate[0], g1 = gate[1], g2 = gate[2];

    // ---- phase A: three gated channels, this wave's 8 N-tiles ----------
#pragma clang loop unroll(disable)
    for (int nt8 = 0; nt8 < 8; ++nt8) {
        const int ntg = w * 8 + nt8;
        const int nc = ntg * 16 + m;
        v8f cN[4], cS[4], cP[4];
#pragma unroll
        for (int t = 0; t < 4; ++t) { cN[t] = vzero(); cS[t] = vzero(); cP[t] = vzero(); }
#pragma unroll
        for (int kc = 0; kc < 16; ++kc) {
            v16bf b = fragB_g(pWnb, kc * 16 + ntg, lane);
#pragma unroll
            for (int t = 0; t < 4; ++t)
                cN[t] = wmma_bf16(fragA_lds(sA1 + t * 16 * ST1, ST1, m, hf, kc), b, cN[t]);
        }
#pragma unroll
        for (int kc = 0; kc < 10; ++kc) {
            v16bf b = fragB_g(pWstr, kc * 16 + ntg, lane);
#pragma unroll
            for (int t = 0; t < 4; ++t)
                cS[t] = wmma_bf16(fragA_lds(sA2 + t * 16 * ST2, ST2, m, hf, kc), b, cS[t]);
        }
        {
            v16bf b = fragB_g(pWpos, ntg, lane);
#pragma unroll
            for (int t = 0; t < 4; ++t)
                cP[t] = wmma_bf16(fragA_lds(sAp + t * 16 * STP, STP, m, hf, 0), b, cP[t]);
        }
        const float bn = b_nb[nc], bp = b_pos[nc], bs = b_str[nc];
#pragma unroll
        for (int t = 0; t < 4; ++t) {
            uint16_t* s3 = sA3 + t * 16 * ST3;
#pragma unroll
            for (int r = 0; r < 8; ++r) {
                float f = g0 * fmaxf(cN[t][r] + bn, 0.f)
                        + g1 * fmaxf(cP[t][r] + bp, 0.f)
                        + g2 * fmaxf(cS[t][r] + bs, 0.f);
                s3[(size_t)(r + hf * 8) * ST3 + nc] = (uint16_t)rne_bf16_bits(f);
            }
        }
    }
    __syncthreads();

    // ---- phase B: relu(h_fused@W1b + b1 + c1) . W2 ---------------------
    float ac[4][8];
#pragma unroll
    for (int t = 0; t < 4; ++t)
#pragma unroll
        for (int r = 0; r < 8; ++r) ac[t][r] = 0.f;

#pragma clang loop unroll(disable)
    for (int nt8 = 0; nt8 < 8; ++nt8) {
        const int ntg = w * 8 + nt8;
        const int nc = ntg * 16 + m;
        v8f c[4];
#pragma unroll
        for (int t = 0; t < 4; ++t) c[t] = vzero();
#pragma unroll
        for (int kc = 0; kc < 8; ++kc) {
            v16bf b = fragB_g(pW1b, kc * 16 + ntg, lane);
#pragma unroll
            for (int t = 0; t < 4; ++t)
                c[t] = wmma_bf16(fragA_lds(sA3 + t * 16 * ST3, ST3, m, hf, kc), b, c[t]);
        }
        const float bb = b1c[nc], w2 = W2[nc];
#pragma unroll
        for (int t = 0; t < 4; ++t)
#pragma unroll
            for (int r = 0; r < 8; ++r) ac[t][r] += fmaxf(c[t][r] + bb, 0.f) * w2;
    }
    // reduce over the 16 N-lanes of each half (xor 1,2,4,8)
#pragma unroll
    for (int mk = 1; mk < 16; mk <<= 1) {
#pragma unroll
        for (int t = 0; t < 4; ++t)
#pragma unroll
            for (int r = 0; r < 8; ++r) ac[t][r] += __shfl_xor(ac[t][r], mk, 32);
    }
    // per-wave partials -> LDS, then cross-wave combine
    if (m < 8) {
        const int row = hf * 8 + m;
#pragma unroll
        for (int t = 0; t < 4; ++t) {
            float v = ac[t][0];
#pragma unroll
            for (int j = 1; j < 8; ++j) if (m == j) v = ac[t][j];
            red[(w * 4 + t) * 16 + row] = v;
        }
    }
    __syncthreads();
    {
        const int t = tidx >> 4, row = tidx & 15;      // 64 threads = 4 tiles x 16 rows
        const int oe = base + t * 16 + row;
        if (oe < E) out[oe] = red[t * 16 + row] + red[(4 + t) * 16 + row] + b2[0];
    }
}

// ======================= host driver ====================================
extern "C" void kernel_launch(void* const* d_in, const int* in_sizes, int n_in,
                              void* d_out, int out_size, void* d_ws, size_t ws_size,
                              hipStream_t stream) {
    (void)n_in; (void)out_size; (void)ws_size;
    const int*   hid   = (const int*)d_in[0];
    const int*   rid   = (const int*)d_in[1];
    const int*   tid   = (const int*)d_in[2];
    const float* q     = (const float*)d_in[3];
    const float* ent   = (const float*)d_in[4];
    const float* rel   = (const float*)d_in[6];
    const float* topic = (const float*)d_in[7];
    const int*   mids  = (const int*)d_in[8];
    const float* mwts  = (const float*)d_in[9];
    const float* ntx   = (const float*)d_in[10];
    const float* memb  = (const float*)d_in[11];
    const float* Wnb   = (const float*)d_in[12]; const float* bnb  = (const float*)d_in[13];
    const float* Wpos  = (const float*)d_in[14]; const float* bpos = (const float*)d_in[15];
    const float* Wstr  = (const float*)d_in[16]; const float* bstr = (const float*)d_in[17];
    const float* Wg    = (const float*)d_in[18]; const float* bg   = (const float*)d_in[19];
    const float* W1    = (const float*)d_in[20]; const float* b1   = (const float*)d_in[21];
    const float* W2    = (const float*)d_in[22]; const float* b2   = (const float*)d_in[23];

    const int E = in_sizes[0];
    const int N = in_sizes[7] / 2;
    const int NTEXT = in_sizes[4] / 256;
    const int R = in_sizes[6] / 256;

    size_t off = 0;
    auto A = [&](size_t bytes) -> char* {
        off = (off + 255) & ~(size_t)255;
        char* p = (char*)d_ws + off; off += bytes; return p;
    };
    float*    cin   = (float*)A((size_t)N * 4);
    float*    cou   = (float*)A((size_t)N * 4);
    float*    pos   = (float*)A((size_t)N * 40);
    float*    cur   = (float*)A((size_t)N * 8);
    float*    nxt   = (float*)A((size_t)N * 8);
    float*    acc   = (float*)A((size_t)N * 8);
    float*    gate  = (float*)A(64);
    float*    b1c   = (float*)A(1024);
    unsigned* pWnb  = (unsigned*)A((size_t)16 * 512 * 8 * 4);
    unsigned* pWstr = (unsigned*)A((size_t)10 * 512 * 8 * 4);
    unsigned* pWpos = (unsigned*)A((size_t)1 * 512 * 8 * 4);
    unsigned* pW1b  = (unsigned*)A((size_t)8 * 512 * 8 * 4);
    uint16_t* motif = (uint16_t*)A((size_t)E * 64 * 2);
    uint16_t* entb  = (uint16_t*)A((size_t)NTEXT * 256 * 2);
    uint16_t* relb  = (uint16_t*)A((size_t)R * 256 * 2);
    uint16_t* ntxb  = (uint16_t*)A(512);

    const int TB = 256;
    auto cdiv = [](long a, long b) { return (int)((a + b - 1) / b); };

    // bf16 pre-conversion of activation tables (halves gather traffic)
    k_cvt<<<cdiv((long)NTEXT * 256, TB), TB, 0, stream>>>(ent, entb, (long)NTEXT * 256);
    k_cvt<<<cdiv((long)R * 256, TB), TB, 0, stream>>>(rel, relb, (long)R * 256);
    k_cvt<<<1, TB, 0, stream>>>(ntx, ntxb, 256);

    // DDE scatter-mean rounds
    k_zero<<<cdiv(N, TB), TB, 0, stream>>>(cin, N);
    k_zero<<<cdiv(N, TB), TB, 0, stream>>>(cou, N);
    k_count<<<cdiv(E, TB), TB, 0, stream>>>(E, hid, tid, cin, cou);
    k_initpos<<<cdiv(N, TB), TB, 0, stream>>>(N, topic, pos, cur);
    float* c = cur; float* x = nxt;
    for (int r = 0; r < 2; ++r) {                         // forward: src=h, dst=t
        k_zero<<<cdiv(2L * N, TB), TB, 0, stream>>>(acc, 2L * N);
        k_scatter<<<cdiv(E, TB), TB, 0, stream>>>(E, c, hid, tid, acc);
        k_norm<<<cdiv(N, TB), TB, 0, stream>>>(N, acc, cin, x, pos, 2 + 2 * r);
        float* tmp = c; c = x; x = tmp;
    }
    k_copy2<<<cdiv(2L * N, TB), TB, 0, stream>>>(N, topic, c);
    for (int r = 0; r < 2; ++r) {                         // reverse: src=t, dst=h
        k_zero<<<cdiv(2L * N, TB), TB, 0, stream>>>(acc, 2L * N);
        k_scatter<<<cdiv(E, TB), TB, 0, stream>>>(E, c, tid, hid, acc);
        k_norm<<<cdiv(N, TB), TB, 0, stream>>>(N, acc, cou, x, pos, 6 + 2 * r);
        float* tmp = c; c = x; x = tmp;
    }
    // motif einsum -> bf16
    k_motif<<<cdiv((long)E * 64, TB), TB, 0, stream>>>((long)E * 64, mids, mwts, memb, motif);
    // weight prepack (bf16 WMMA B-fragment layout)
    k_prepack<<<cdiv(16 * 512, TB), TB, 0, stream>>>(Wnb, 512, 16, pWnb);
    k_prepack<<<cdiv(10 * 512, TB), TB, 0, stream>>>(Wstr, 320, 10, pWstr);
    k_prepack<<<cdiv(1 * 512, TB), TB, 0, stream>>>(Wpos, 20, 1, pWpos);
    k_prepack<<<cdiv(8 * 512, TB), TB, 0, stream>>>(W1 + 256 * 256, 256, 8, pW1b);
    // gate + hoisted q@W1 top half
    k_gate_c1<<<1, 256, 0, stream>>>(q, W1, b1, Wg, bg, b1c, gate);
    // fused WMMA edge kernel: 64 edges per block (2 waves), dynamic LDS
    const size_t smem_bytes = (size_t)SMEM_ELTS * 2;     // 146,944 B
    k_edges<<<cdiv(E, 64), 64, smem_bytes, stream>>>(E, NTEXT, hid, tid, rid,
                                                     entb, ntxb, relb,
                                                     motif, pos, gate, bnb, bpos, bstr, b1c, W2, b2,
                                                     pWnb, pWstr, pWpos, pW1b, (float*)d_out);
}
